// InfraredFeatureExtractor_11227044512391
// MI455X (gfx1250) — matrix operand order repaired and checked by reading, not compiled, not run on hardware
//
#include <hip/hip_runtime.h>
#include <cstdint>

typedef __attribute__((ext_vector_type(16))) _Float16 v16h;
typedef __attribute__((ext_vector_type(8)))  _Float16 v8h;
typedef __attribute__((ext_vector_type(8)))  float    v8f;

#define HH 256
#define WW 256
#define NN (HH*WW)
#define BB 2
#define C0 16
#define C1 32
#define C2 64
#define RR (BB*NN)   // 131072 node-rows total
#define SP 32        // padded NHWC stride (halves) for GAT inputs

// ---------------------------------------------------------------- conv1 5x5, 1->16, +bias, relu
__global__ void k_conv1(const float* __restrict__ x, const float* __restrict__ w,
                        const float* __restrict__ bias, float* __restrict__ out) {
  int i = blockIdx.x * blockDim.x + threadIdx.x;          // B*C0*N
  int p = i % NN; int c = (i / NN) % C0; int b = i / (NN * C0);
  int y = p / WW, xx = p % WW;
  float acc = bias[c];
  for (int ky = 0; ky < 5; ky++) {
    int yy = y + ky - 2; if (yy < 0 || yy >= HH) continue;
    for (int kx = 0; kx < 5; kx++) {
      int xc = xx + kx - 2; if (xc < 0 || xc >= WW) continue;
      acc += x[b * NN + yy * WW + xc] * w[c * 25 + ky * 5 + kx];
    }
  }
  out[i] = fmaxf(acc, 0.f);
}

// ---------------------------------------------------------------- per-channel mean/rstd (biased var)
__global__ void k_bn_stats(const float* __restrict__ x, float* __restrict__ stats, int C) {
  int c = blockIdx.x;
  __shared__ float ssum[256], ssq[256];
  float s = 0.f, q = 0.f;
  for (int b = 0; b < BB; b++) {
    const float* p = x + (size_t)(b * C + c) * NN;
    for (int i = threadIdx.x; i < NN; i += blockDim.x) { float v = p[i]; s += v; q += v * v; }
  }
  ssum[threadIdx.x] = s; ssq[threadIdx.x] = q; __syncthreads();
  for (int off = 128; off > 0; off >>= 1) {
    if ((int)threadIdx.x < off) { ssum[threadIdx.x] += ssum[threadIdx.x + off]; ssq[threadIdx.x] += ssq[threadIdx.x + off]; }
    __syncthreads();
  }
  if (threadIdx.x == 0) {
    float inv = 1.f / (float)(BB * NN);
    float mu = ssum[0] * inv;
    float var = ssq[0] * inv - mu * mu;
    stats[2 * c] = mu; stats[2 * c + 1] = rsqrtf(fmaxf(var, 0.f) + 1e-5f);
  }
}

// ---------------------------------------------------------------- bn apply (NCHW f32 -> NCHW f32)
__global__ void k_bn_apply(const float* __restrict__ x, const float* __restrict__ stats,
                           const float* __restrict__ g, const float* __restrict__ bt,
                           float* __restrict__ out, int C) {
  int i = blockIdx.x * blockDim.x + threadIdx.x;          // B*C*N
  int c = (i / NN) % C;
  out[i] = g[c] * (x[i] - stats[2 * c]) * stats[2 * c + 1] + bt[c];
}

// ---------------------------------------------------------------- bn + positional encoding -> NHWC f16 (stride SP, zero-padded)
__global__ void k_bn_pe_nhwc(const float* __restrict__ x, const float* __restrict__ stats,
                             const float* __restrict__ g, const float* __restrict__ bt,
                             const float* __restrict__ pw, const float* __restrict__ pb,
                             _Float16* __restrict__ out, int C) {
  int i = blockIdx.x * blockDim.x + threadIdx.x;          // B*N*SP
  int c = i % SP; int p = (i / SP) % NN; int b = i / (SP * NN);
  float v = 0.f;
  if (c < C) {
    int y = p / WW, xx = p % WW;
    float py = -1.f + 2.f * (float)y  / (float)(HH - 1);
    float px = -1.f + 2.f * (float)xx / (float)(WW - 1);
    v = x[(size_t)(b * C + c) * NN + p];
    v = g[c] * (v - stats[2 * c]) * stats[2 * c + 1] + bt[c];
    v += pw[c * 2 + 0] * py + pw[c * 2 + 1] * px + pb[c];
  }
  out[i] = (_Float16)v;
}

// ---------------------------------------------------------------- pack weight (NC x KR f32) into f16 WMMA B-tiles
// layout: tile nt, lane, j  ->  B[k][n], n = nt*16 + (lane&15), k = j + (lane<16?0:16)
__global__ void k_pack_w(const float* __restrict__ Wm, _Float16* __restrict__ out,
                         int KR, int NC) {
  int idx = blockIdx.x * blockDim.x + threadIdx.x;
  int total = (NC / 16) * 32 * 16;
  if (idx >= total) return;
  int j = idx & 15; int lane = (idx >> 4) & 31; int nt = idx >> 9;
  int n = nt * 16 + (lane & 15);
  int k = j + ((lane < 16) ? 0 : 16);
  out[idx] = (k < KR) ? (_Float16)Wm[n * KR + k] : (_Float16)0.f;
}

// ---------------------------------------------------------------- node-feature GEMM: h = xg @ W^T  (A from padded stride-32 f16)
template <int NC>
__global__ void k_gemm(const _Float16* __restrict__ xg, const v16h* __restrict__ bpack,
                       float* __restrict__ hout) {
  int wave = (blockIdx.x * blockDim.x + threadIdx.x) >> 5;  // RR/16 waves
  int lane = threadIdx.x & 31;
  int row0 = wave * 16;
  const _Float16* row = xg + (size_t)(row0 + (lane & 15)) * SP;
  int kofs = (lane < 16) ? 0 : 16;
  v8h lo = *(const v8h*)(row + kofs);
  v8h hi = *(const v8h*)(row + kofs + 8);
  v16h a;
#pragma unroll
  for (int j = 0; j < 8; j++) { a[j] = lo[j]; a[j + 8] = hi[j]; }
  int mbase = (lane < 16) ? 0 : 8;
#pragma unroll
  for (int nt = 0; nt < NC / 16; nt++) {
    v16h bv = bpack[nt * 32 + lane];
    v8f c = {};
    c = __builtin_amdgcn_wmma_f32_16x16x32_f16(false, a, false, bv, (short)0, c, false, false);
    int n = nt * 16 + (lane & 15);
#pragma unroll
    for (int r = 0; r < 8; r++)
      hout[(size_t)(row0 + mbase + r) * NC + n] = c[r];
  }
}

// ---------------------------------------------------------------- attention scores s=h.as, d=h.ad
__global__ void k_scores(const float* __restrict__ h, const float* __restrict__ as,
                         const float* __restrict__ ad, float* __restrict__ s,
                         float* __restrict__ d, int NC) {
  int i = blockIdx.x * blockDim.x + threadIdx.x;           // RR
  const float* hp = h + (size_t)i * NC;
  float ss = 0.f, dd = 0.f;
  for (int c = 0; c < NC; c++) { float v = hp[c]; ss += v * as[c]; dd += v * ad[c]; }
  s[i] = ss; d[i] = dd;
}

// ---------------------------------------------------------------- GAT aggregate (wave per dst node), relu'd NCHW out
template <int NC>
__global__ void k_gat(const float* __restrict__ h, const float* __restrict__ s,
                      const float* __restrict__ dsc, const float* __restrict__ gb,
                      float* __restrict__ out) {
  int gw = (blockIdx.x * blockDim.x + threadIdx.x) >> 5;   // node in [0,RR)
  int lane = threadIdx.x & 31;
  int b = gw / NN; int p = gw % NN; int y = p / WW, x = p % WW;
  float dv = dsc[gw];
  float e[9]; int sp[9]; bool ok[9];
#pragma unroll
  for (int t = 0; t < 9; t++) {
    int dy = t / 3 - 1, dx = t % 3 - 1;
    int yy = y + dy, xx = x + dx;
    bool v = (yy >= 0 && yy < HH && xx >= 0 && xx < WW);
    ok[t] = v;
    sp[t] = v ? (b * NN + yy * WW + xx) : gw;
    float ev = s[sp[t]] + dv;                              // safe addr even if !ok
    ev = (ev > 0.f) ? ev : 0.2f * ev;
    e[t] = v ? ev : -1e30f;
  }
  float m = -1e30f;
#pragma unroll
  for (int t = 0; t < 9; t++) m = fmaxf(m, e[t]);
  float pz[9]; float denom = 0.f;
#pragma unroll
  for (int t = 0; t < 9; t++) { float pv = ok[t] ? expf(e[t] - m) : 0.f; pz[t] = pv; denom += pv; }
  float inv = 1.f / (denom + 1e-16f);
  for (int c = lane; c < NC; c += 32) {
    float acc = 0.f;
#pragma unroll
    for (int t = 0; t < 9; t++) acc += pz[t] * h[(size_t)sp[t] * NC + c];
    float v = acc * inv + gb[c];
    out[((size_t)(b * NC + c)) * NN + p] = fmaxf(v, 0.f);
  }
}

// ---------------------------------------------------------------- global mean pool per (b,c)
__global__ void k_pool(const float* __restrict__ f, float* __restrict__ pooled) {
  int bc = blockIdx.x;
  __shared__ float sm[256];
  const float* p = f + (size_t)bc * NN;
  float s = 0.f;
  for (int i = threadIdx.x; i < NN; i += 256) s += p[i];
  sm[threadIdx.x] = s; __syncthreads();
  for (int off = 128; off > 0; off >>= 1) {
    if ((int)threadIdx.x < off) sm[threadIdx.x] += sm[threadIdx.x + off];
    __syncthreads();
  }
  if (threadIdx.x == 0) pooled[bc] = sm[0] / (float)NN;
}

// ---------------------------------------------------------------- channel-attention MLP + softmax
__global__ void k_mlp(const float* __restrict__ pooled, const float* __restrict__ w1,
                      const float* __restrict__ b1, const float* __restrict__ w2,
                      const float* __restrict__ b2, float* __restrict__ cw) {
  int b = blockIdx.x, t = threadIdx.x;                     // 64 threads
  __shared__ float hs[16]; __shared__ float lg[64]; __shared__ float ms, ssum;
  if (t < 16) {
    float a = b1[t];
    for (int k = 0; k < C2; k++) a += pooled[b * C2 + k] * w1[t * C2 + k];
    hs[t] = fmaxf(a, 0.f);
  }
  __syncthreads();
  { float l = b2[t]; for (int k = 0; k < 16; k++) l += hs[k] * w2[t * 16 + k]; lg[t] = l; }
  __syncthreads();
  if (t == 0) {
    float m = -1e30f; for (int k = 0; k < C2; k++) m = fmaxf(m, lg[k]);
    float s = 0.f;    for (int k = 0; k < C2; k++) s += expf(lg[k] - m);
    ms = m; ssum = s;
  }
  __syncthreads();
  cw[b * C2 + t] = expf(lg[t] - ms) / ssum;
}

// ---------------------------------------------------------------- ff = f*cw*filt + filt_bias (NCHW f32 + NHWC f16)
__global__ void k_ff(const float* __restrict__ f3, const float* __restrict__ cw,
                     const float* __restrict__ filt, const float* __restrict__ fb,
                     float* __restrict__ ffn, _Float16* __restrict__ ffh) {
  int i = blockIdx.x * blockDim.x + threadIdx.x;           // B*C2*N
  int p = i % NN; int c = (i / NN) % C2; int b = i / (NN * C2);
  float v = f3[i] * cw[b * C2 + c] * filt[(size_t)c * NN + p] + fb[c];
  ffn[i] = v;
  ffh[((size_t)(b * NN + p)) * C2 + c] = (_Float16)v;
}

// ---------------------------------------------------------------- 3x3 conv via implicit-GEMM WMMA over 9 taps
// TO_NHWC: bias+relu -> NHWC f16   |  else: bias + ff residual + relu -> NCHW f32
template <int CIN, int COUT, bool TO_NHWC>
__global__ void k_conv3(const _Float16* __restrict__ in, const float* __restrict__ w,
                        const float* __restrict__ bias, _Float16* __restrict__ outh,
                        float* __restrict__ outf, const float* __restrict__ ffn) {
  constexpr int CHUNKS = CIN / 32;
  constexpr int CT = COUT / 16;
  __shared__ v16h bsm[9 * CHUNKS * CT * 32];               // 36 KB for both instantiations
  _Float16* bs = (_Float16*)bsm;
  const int total = 9 * CHUNKS * CT * 32 * 16;
  for (int idx = threadIdx.x; idx < total; idx += blockDim.x) {
    int j    = idx & 15;
    int lane = (idx >> 4) & 31;
    int rest = idx >> 9;
    int ct    = rest % CT;
    int chunk = (rest / CT) % CHUNKS;
    int tap   = rest / (CT * CHUNKS);
    int co = ct * 16 + (lane & 15);
    int ci = chunk * 32 + j + ((lane < 16) ? 0 : 16);
    bs[idx] = (_Float16)w[(co * CIN + ci) * 9 + tap];
  }
  __syncthreads();

  int wave = threadIdx.x >> 5, lane = threadIdx.x & 31;
  int tile = blockIdx.x * 8 + wave;                        // 16-pixel row strip
  int pix0 = tile * 16;
  int b  = pix0 / NN;
  int p0 = pix0 % NN;
  int y  = p0 / WW;
  int x0 = p0 % WW;
  int xs0 = x0 + (lane & 15);
  int kofs = (lane < 16) ? 0 : 16;
  const v8h vz = {};

  v8f acc[CT] = {};
#pragma unroll
  for (int tap = 0; tap < 9; tap++) {
    int dy = tap / 3 - 1, dx = tap % 3 - 1;
    int yy = y + dy, xx = xs0 + dx;
    bool valid = (yy >= 0 && yy < HH && xx >= 0 && xx < WW);
    int yc = yy < 0 ? 0 : (yy >= HH ? HH - 1 : yy);        // clamped, always in-bounds
    int xc = xx < 0 ? 0 : (xx >= WW ? WW - 1 : xx);
    const _Float16* rowp = in + ((size_t)(b * NN + yc * WW + xc)) * CIN;
#pragma unroll
    for (int chunk = 0; chunk < CHUNKS; chunk++) {
      v8h lo = *(const v8h*)(rowp + chunk * 32 + kofs);
      v8h hi = *(const v8h*)(rowp + chunk * 32 + kofs + 8);
      lo = valid ? lo : vz;                                // vector select, no exec toggling
      hi = valid ? hi : vz;
      v16h a;
#pragma unroll
      for (int j = 0; j < 8; j++) { a[j] = lo[j]; a[j + 8] = hi[j]; }
#pragma unroll
      for (int ct = 0; ct < CT; ct++) {
        v16h bv = bsm[((tap * CHUNKS + chunk) * CT + ct) * 32 + lane];
        acc[ct] = __builtin_amdgcn_wmma_f32_16x16x32_f16(false, a, false, bv, (short)0, acc[ct], false, false);
      }
    }
  }
  int mbase = (lane < 16) ? 0 : 8;
#pragma unroll
  for (int ct = 0; ct < CT; ct++) {
    int co = ct * 16 + (lane & 15);
#pragma unroll
    for (int r = 0; r < 8; r++) {
      int pg = pix0 + mbase + r;
      float v = acc[ct][r] + bias[co];
      if (TO_NHWC) {
        outh[(size_t)pg * COUT + co] = (_Float16)fmaxf(v, 0.f);
      } else {
        int bb = pg / NN; int pp = pg % NN;
        size_t oi = ((size_t)(bb * COUT + co)) * NN + pp;
        v += ffn[oi];
        outf[oi] = fmaxf(v, 0.f);
      }
    }
  }
}

// ================================================================ launch
extern "C" void kernel_launch(void* const* d_in, const int* in_sizes, int n_in,
                              void* d_out, int out_size, void* d_ws, size_t ws_size,
                              hipStream_t stream) {
  (void)in_sizes; (void)n_in; (void)out_size; (void)ws_size;
  const float* x       = (const float*)d_in[0];
  const float* conv1_w = (const float*)d_in[1];
  const float* conv1_b = (const float*)d_in[2];
  const float* bn1_g   = (const float*)d_in[3];
  const float* bn1_b   = (const float*)d_in[4];
  const float* pos1_w  = (const float*)d_in[5];
  const float* pos1_b  = (const float*)d_in[6];
  const float* gat1_w  = (const float*)d_in[7];
  const float* gat1_as = (const float*)d_in[8];
  const float* gat1_ad = (const float*)d_in[9];
  const float* gat1_b  = (const float*)d_in[10];
  const float* bn2_g   = (const float*)d_in[11];
  const float* bn2_b   = (const float*)d_in[12];
  const float* pos2_w  = (const float*)d_in[13];
  const float* pos2_b  = (const float*)d_in[14];
  const float* gat2_w  = (const float*)d_in[15];
  const float* gat2_as = (const float*)d_in[16];
  const float* gat2_ad = (const float*)d_in[17];
  const float* gat2_b  = (const float*)d_in[18];
  const float* bn3_g   = (const float*)d_in[19];
  const float* bn3_b   = (const float*)d_in[20];
  const float* mlp1_w  = (const float*)d_in[21];
  const float* mlp1_b  = (const float*)d_in[22];
  const float* mlp2_w  = (const float*)d_in[23];
  const float* mlp2_b  = (const float*)d_in[24];
  const float* filt    = (const float*)d_in[25];
  const float* filt_b  = (const float*)d_in[26];
  const float* enh1_w  = (const float*)d_in[27];
  const float* enh1_b  = (const float*)d_in[28];
  const float* enh2_w  = (const float*)d_in[29];
  const float* enh2_b  = (const float*)d_in[30];
  const float* bn4_g   = (const float*)d_in[31];
  const float* bn4_b   = (const float*)d_in[32];
  float* outp = (float*)d_out;

  uint8_t* base = (uint8_t*)d_ws;
  size_t off = 0;
  auto take = [&](size_t bytes) -> void* {
    void* p = base + off;
    off = (off + bytes + 255) & ~(size_t)255;
    return p;
  };
  float*    bufA   = (float*)take((size_t)RR * 64 * 4);    // NCHW f32
  float*    bufB   = (float*)take((size_t)RR * 64 * 4);    // NCHW f32
  float*    bufC   = (float*)take((size_t)RR * 64 * 4);    // h (node-major) f32
  _Float16* bufD   = (_Float16*)take((size_t)RR * 64 * 2); // NHWC f16 (also stride-32 GAT input)
  _Float16* bufE   = (_Float16*)take((size_t)RR * 32 * 2); // NHWC f16
  float*    sbuf   = (float*)take((size_t)RR * 4);
  float*    dbuf   = (float*)take((size_t)RR * 4);
  float*    stats  = (float*)take(2 * 64 * 4);
  float*    pooled = (float*)take(BB * 64 * 4);
  float*    cw     = (float*)take(BB * 64 * 4);
  _Float16* wpk1   = (_Float16*)take((C1 / 16) * 32 * 16 * 2);
  _Float16* wpk2   = (_Float16*)take((C2 / 16) * 32 * 16 * 2);

  // ---- pack GAT weights into WMMA B-tile f16 layout (one-shot, tiny)
  k_pack_w<<<4, 256, 0, stream>>>(gat1_w, wpk1, C0, C1);
  k_pack_w<<<8, 256, 0, stream>>>(gat2_w, wpk2, C1, C2);

  // ---- stage 1: conv1 + relu + bn1
  k_conv1<<<RR * C0 / 256, 256, 0, stream>>>(x, conv1_w, conv1_b, bufA);
  k_bn_stats<<<C0, 256, 0, stream>>>(bufA, stats, C0);
  k_bn_pe_nhwc<<<RR * SP / 256, 256, 0, stream>>>(bufA, stats, bn1_g, bn1_b, pos1_w, pos1_b, bufD, C0);

  // ---- GAT block 1 (WMMA projection)
  k_gemm<C1><<<RR / 16 / 8, 256, 0, stream>>>(bufD, (const v16h*)wpk1, bufC);
  k_scores<<<RR / 256, 256, 0, stream>>>(bufC, gat1_as, gat1_ad, sbuf, dbuf, C1);
  k_gat<C1><<<RR * 32 / 256, 256, 0, stream>>>(bufC, sbuf, dbuf, gat1_b, bufA);
  k_bn_stats<<<C1, 256, 0, stream>>>(bufA, stats, C1);
  k_bn_pe_nhwc<<<RR * SP / 256, 256, 0, stream>>>(bufA, stats, bn2_g, bn2_b, pos2_w, pos2_b, bufD, C1);

  // ---- GAT block 2 (WMMA projection)
  k_gemm<C2><<<RR / 16 / 8, 256, 0, stream>>>(bufD, (const v16h*)wpk2, bufC);
  k_scores<<<RR / 256, 256, 0, stream>>>(bufC, gat2_as, gat2_ad, sbuf, dbuf, C2);
  k_gat<C2><<<RR * 32 / 256, 256, 0, stream>>>(bufC, sbuf, dbuf, gat2_b, bufA);
  k_bn_stats<<<C2, 256, 0, stream>>>(bufA, stats, C2);
  k_bn_apply<<<RR * C2 / 256, 256, 0, stream>>>(bufA, stats, bn3_g, bn3_b, bufB, C2);

  // ---- channel attention + filtered features
  k_pool<<<BB * C2, 256, 0, stream>>>(bufB, pooled);
  k_mlp<<<BB, 64, 0, stream>>>(pooled, mlp1_w, mlp1_b, mlp2_w, mlp2_b, cw);
  k_ff<<<RR * C2 / 256, 256, 0, stream>>>(bufB, cw, filt, filt_b, bufA, bufD);

  // ---- enhancement convs via WMMA implicit GEMM, residual, bn4
  k_conv3<C2, C1, true ><<<RR / 16 / 8, 256, 0, stream>>>(bufD, enh1_w, enh1_b, bufE, nullptr, nullptr);
  k_conv3<C1, C2, false><<<RR / 16 / 8, 256, 0, stream>>>(bufE, enh2_w, enh2_b, nullptr, bufB, bufA);
  k_bn_stats<<<C2, 256, 0, stream>>>(bufB, stats, C2);
  k_bn_apply<<<RR * C2 / 256, 256, 0, stream>>>(bufB, stats, bn4_g, bn4_b, outp, C2);
}
